// CausalMultiHeadAttention_14594298872006
// MI455X (gfx1250) — matrix operand order repaired and checked by reading, not compiled, not run on hardware
//
#include <hip/hip_runtime.h>
#include <hip/hip_bf16.h>

typedef __bf16 bf16_t;
typedef __attribute__((ext_vector_type(16))) __bf16 v16bf;
typedef __attribute__((ext_vector_type(8)))  float  v8f;

union FragBF {
    v16bf  v;
    uint4  u[2];
    bf16_t e[16];
};

#define WMMA_BF16(Afrag, Bfrag, Cacc) \
    __builtin_amdgcn_wmma_f32_16x16x32_bf16(false, (Afrag), false, (Bfrag), (short)0, (Cacc), false, false)

// ---------------------------------------------------------------------------
// Problem constants
// ---------------------------------------------------------------------------
#define BB   8
#define TT   1024
#define CC   576
#define NH   9
#define NKV  3
#define HD   64
#define KVC  192
#define MM   (BB * TT)          // 8192 rows

// Workspace layout (bytes, all 256-aligned)
#define OFF_XB   ((size_t)0)                      // x bf16      [8192 x 576]
#define OFF_WQT  ((size_t)9437184)                // Wq^T bf16   [576 x 576]
#define OFF_WKT  ((size_t)10100736)               // Wk^T bf16   [192 x 576]
#define OFF_WVT  ((size_t)10321920)               // Wv^T bf16   [192 x 576]
#define OFF_WOT  ((size_t)10543104)               // Wo^T bf16   [576 x 576]
#define OFF_QB   ((size_t)11206656)               // q bf16      [8192 x 576]
#define OFF_KB   ((size_t)20643840)               // k bf16      [8192 x 192]
#define OFF_VT   ((size_t)23789568)               // v^T bf16    [8][3][64][1024]
#define OFF_YB   ((size_t)26935296)               // y bf16      [8192 x 576]

// ---------------------------------------------------------------------------
// Stage 0: fp32 -> bf16 convert / weight transpose
// ---------------------------------------------------------------------------
__global__ void cvt_bf16_kernel(const float* __restrict__ in,
                                bf16_t* __restrict__ out, int n) {
    int i = blockIdx.x * blockDim.x + threadIdx.x;
    if (i < n) out[i] = (bf16_t)in[i];
}

// W is [K x N] row-major (input-dim major); produce Wt[n][k] bf16
__global__ void transpose_bf16_kernel(const float* __restrict__ W,
                                      bf16_t* __restrict__ Wt, int K, int N) {
    int i = blockIdx.x * blockDim.x + threadIdx.x;
    if (i < K * N) {
        int n = i % N;
        int k = i / N;
        Wt[(size_t)n * K + k] = (bf16_t)W[i];
    }
}

// ---------------------------------------------------------------------------
// Stage 1/3: WMMA GEMM  C[M,N] = A[M,K] @ W + bias
//   A: bf16 row-major.  Bt: bf16 [N][K] (pre-transposed weights).
//   mode 0: bf16 row-major out.  mode 1: bf16 transposed out [b][n][t] (for V).
//   mode 2: fp32 row-major out (final projection).
//   block = 128 threads (4 waves); wave -> 32x64 tile (2 M-subtiles x 4 N-tiles,
//   8 accumulators).  Each B fragment feeds 2 wmmas, each A fragment 4.
//   Block M-tile = 128 rows.
// ---------------------------------------------------------------------------
__global__ __launch_bounds__(128)
void gemm_wmma_kernel(const bf16_t* __restrict__ A, const bf16_t* __restrict__ Bt,
                      const float* __restrict__ bias, void* __restrict__ Cout,
                      int M, int N, int K, int mode) {
    const int  lane = threadIdx.x & 31;
    const int  wave = threadIdx.x >> 5;
    const int  n16  = lane & 15;
    const bool hi   = lane >= 16;
    const int  mb   = blockIdx.x * 128 + wave * 32;   // wave owns rows mb..mb+31
    const int  nb   = blockIdx.y * 64;

    const int gA = hi ? 8 : 0;     // A-fragment K sub-group
    const int kB = hi ? 16 : 0;    // unused; B fragments use gA like A

    const bf16_t* Arow0 = A + (size_t)(mb + n16) * K;        // M-subtile 0
    const bf16_t* Arow1 = Arow0 + (size_t)16 * K;            // M-subtile 1

    v8f acc[2][4];
#pragma unroll
    for (int s = 0; s < 2; ++s)
#pragma unroll
        for (int t = 0; t < 4; ++t)
#pragma unroll
            for (int r = 0; r < 8; ++r) acc[s][t][r] = 0.0f;

    for (int k = 0; k < K; k += 32) {
        __builtin_prefetch(Arow0 + k + 64, 0, 1);   // global_prefetch_b8
        __builtin_prefetch(Arow1 + k + 64, 0, 1);
        FragBF a0, a1;
        a0.u[0] = *(const uint4*)(Arow0 + k + gA);
        a0.u[1] = *(const uint4*)(Arow0 + k + 16 + gA);
        a1.u[0] = *(const uint4*)(Arow1 + k + gA);
        a1.u[1] = *(const uint4*)(Arow1 + k + 16 + gA);
#pragma unroll
        for (int t = 0; t < 4; ++t) {
            const bf16_t* Bp = Bt + (size_t)(nb + t * 16 + n16) * K + k + gA;
            FragBF bfr;
            bfr.u[0] = ((const uint4*)Bp)[0];
            bfr.u[1] = ((const uint4*)(Bp + 16))[0];
            acc[0][t] = WMMA_BF16(a0.v, bfr.v, acc[0][t]);
            acc[1][t] = WMMA_BF16(a1.v, bfr.v, acc[1][t]);
        }
    }

#pragma unroll
    for (int s = 0; s < 2; ++s) {
#pragma unroll
        for (int t = 0; t < 4; ++t) {
            const int   col = nb + t * 16 + n16;
            const float bv  = bias[col];
#pragma unroll
            for (int r = 0; r < 8; ++r) {
                const int   m   = mb + s * 16 + (hi ? 8 : 0) + r;
                const float val = acc[s][t][r] + bv;
                if (mode == 0) {
                    ((bf16_t*)Cout)[(size_t)m * N + col] = (bf16_t)val;
                } else if (mode == 1) {
                    // V^T layout: [b][col(=kvh*64+d)][t]   (col < 192)
                    const int b  = m >> 10;          // m / 1024
                    const int tt = m & 1023;
                    ((bf16_t*)Cout)[(size_t)b * (KVC * TT) + (size_t)col * TT + tt] = (bf16_t)val;
                } else {
                    ((float*)Cout)[(size_t)m * N + col] = val;
                }
            }
        }
    }
}

// ---------------------------------------------------------------------------
// Stage 2: causal flash attention, transposed formulation.
//   S^T = K·Q^T (16 keys x 16 q per tile), O^T = V^T·P^T.
//   Lane owns one query column -> softmax stats are lane-local up to one
//   shfl_xor(16) merge.  grid = (B*NH, T/64), block = 128 (4 waves x 16 q).
// ---------------------------------------------------------------------------
__global__ __launch_bounds__(128)
void attn_wmma_kernel(const bf16_t* __restrict__ Q, const bf16_t* __restrict__ Kb,
                      const bf16_t* __restrict__ Vt, bf16_t* __restrict__ Y) {
    const int  lane = threadIdx.x & 31;
    const int  wave = threadIdx.x >> 5;
    const int  n16  = lane & 15;
    const bool hi   = lane >= 16;

    const int bh  = blockIdx.x;
    const int b   = bh / NH;
    const int h   = bh % NH;
    const int kvh = h / (NH / NKV);
    const int qb    = blockIdx.y * 64 + wave * 16;
    const int q_idx = qb + n16;

    const int gA = hi ? 8 : 0;

    // Q^T B-fragments (contraction d = 0..31, 32..63)
    FragBF qf[2];
    {
        const bf16_t* qrow = Q + (size_t)(b * TT + q_idx) * CC + h * HD;
#pragma unroll
        for (int s = 0; s < 2; ++s) {
            const bf16_t* p = qrow + s * 32 + gA;
            qf[s].u[0] = ((const uint4*)p)[0];
            qf[s].u[1] = ((const uint4*)(p + 16))[0];
        }
    }

    const bf16_t* Kbase = Kb + (size_t)(b * TT) * KVC + kvh * HD;
    const bf16_t* Vbase = Vt + (size_t)(b * NKV + kvh) * (HD * TT);

    float m_run = -3.0e38f;
    float l_run = 0.0f;
    v8f   o[4];
#pragma unroll
    for (int t = 0; t < 4; ++t)
#pragma unroll
        for (int r = 0; r < 8; ++r) o[t][r] = 0.0f;

    const int kend = ((qb + 16 + 31) >> 5) << 5;   // causal bound, multiple of 32

    for (int kb = 0; kb < kend; kb += 32) {
        // ---- S^T = K @ Q^T : two 16x16 key tiles, contraction d=64 ----
        v8f s0, s1;
#pragma unroll
        for (int r = 0; r < 8; ++r) { s0[r] = 0.0f; s1[r] = 0.0f; }
#pragma unroll
        for (int ds = 0; ds < 2; ++ds) {
            FragBF ka;
            const bf16_t* kr0 = Kbase + (size_t)(kb + n16) * KVC + ds * 32 + gA;
            ka.u[0] = ((const uint4*)kr0)[0];
            ka.u[1] = ((const uint4*)(kr0 + 16))[0];
            s0 = WMMA_BF16(ka.v, qf[ds].v, s0);
            const bf16_t* kr1 = Kbase + (size_t)(kb + 16 + n16) * KVC + ds * 32 + gA;
            ka.u[0] = ((const uint4*)kr1)[0];
            ka.u[1] = ((const uint4*)(kr1 + 16))[0];
            s1 = WMMA_BF16(ka.v, qf[ds].v, s1);
        }

        // ---- mask + scale + online softmax (lane-local + one half-merge) ----
        float p0[8], p1[8];
        float mx = -3.0e38f;
        const int k0base = kb + gA;
#pragma unroll
        for (int r = 0; r < 8; ++r) {
            float a0 = s0[r] * 0.125f;   // 1/sqrt(64)
            float a1 = s1[r] * 0.125f;
            if (k0base + r > q_idx)      a0 = -3.0e38f;
            if (k0base + 16 + r > q_idx) a1 = -3.0e38f;
            p0[r] = a0;
            p1[r] = a1;
            mx = fmaxf(mx, fmaxf(a0, a1));
        }
        mx = fmaxf(mx, __shfl_xor(mx, 16, 32));
        const float mnew  = fmaxf(m_run, mx);
        const float alpha = __expf(m_run - mnew);
        float rs = 0.0f;
#pragma unroll
        for (int r = 0; r < 8; ++r) {
            p0[r] = __expf(p0[r] - mnew);
            p1[r] = __expf(p1[r] - mnew);
            rs += p0[r] + p1[r];
        }
        rs += __shfl_xor(rs, 16, 32);
        l_run = l_run * alpha + rs;
        m_run = mnew;

#pragma unroll
        for (int t = 0; t < 4; ++t)
#pragma unroll
            for (int r = 0; r < 8; ++r) o[t][r] *= alpha;

        // ---- P^T C layout already matches the B operand: pack lane-locally ----
        FragBF pf;
#pragma unroll
        for (int r = 0; r < 8; ++r) {
            const float q0 = __shfl_xor(p0[r], 16, 32);
            const float q1 = __shfl_xor(p1[r], 16, 32);
            pf.e[r]     = (bf16_t)p0[r];
            pf.e[8 + r] = (bf16_t)p1[r];
        }

        // ---- O^T += V^T @ P^T : 4 d-tiles, contraction keys=32 ----
#pragma unroll
        for (int t = 0; t < 4; ++t) {
            FragBF vf;
            const bf16_t* vr = Vbase + (size_t)(t * 16 + n16) * TT + kb + gA;
            vf.u[0] = ((const uint4*)vr)[0];
            vf.u[1] = ((const uint4*)(vr + 16))[0];
            o[t] = WMMA_BF16(vf.v, pf.v, o[t]);
        }
    }

    const float inv_l = 1.0f / l_run;
    bf16_t* yrow = Y + (size_t)(b * TT + q_idx) * CC + h * HD;
#pragma unroll
    for (int t = 0; t < 4; ++t)
#pragma unroll
        for (int r = 0; r < 8; ++r) {
            const int d = t * 16 + gA + r;
            yrow[d] = (bf16_t)(o[t][r] * inv_l);
        }
}

// ---------------------------------------------------------------------------
// Host launcher
// ---------------------------------------------------------------------------
extern "C" void kernel_launch(void* const* d_in, const int* in_sizes, int n_in,
                              void* d_out, int out_size, void* d_ws, size_t ws_size,
                              hipStream_t stream) {
    const float* x  = (const float*)d_in[0];
    const float* Wq = (const float*)d_in[1];
    const float* bq = (const float*)d_in[2];
    const float* Wk = (const float*)d_in[3];
    const float* bk = (const float*)d_in[4];
    const float* Wv = (const float*)d_in[5];
    const float* bv = (const float*)d_in[6];
    const float* Wo = (const float*)d_in[7];
    const float* bo = (const float*)d_in[8];
    float* out = (float*)d_out;

    char* ws = (char*)d_ws;
    bf16_t* xb  = (bf16_t*)(ws + OFF_XB);
    bf16_t* wqt = (bf16_t*)(ws + OFF_WQT);
    bf16_t* wkt = (bf16_t*)(ws + OFF_WKT);
    bf16_t* wvt = (bf16_t*)(ws + OFF_WVT);
    bf16_t* wot = (bf16_t*)(ws + OFF_WOT);
    bf16_t* qb  = (bf16_t*)(ws + OFF_QB);
    bf16_t* kb  = (bf16_t*)(ws + OFF_KB);
    bf16_t* vt  = (bf16_t*)(ws + OFF_VT);
    bf16_t* yb  = (bf16_t*)(ws + OFF_YB);

    // Stage 0: converts
    {
        const int nx = MM * CC;                       // 4,718,592
        cvt_bf16_kernel<<<(nx + 255) / 256, 256, 0, stream>>>(x, xb, nx);
        transpose_bf16_kernel<<<(CC * CC + 255) / 256, 256, 0, stream>>>(Wq, wqt, CC, CC);
        transpose_bf16_kernel<<<(CC * KVC + 255) / 256, 256, 0, stream>>>(Wk, wkt, CC, KVC);
        transpose_bf16_kernel<<<(CC * KVC + 255) / 256, 256, 0, stream>>>(Wv, wvt, CC, KVC);
        transpose_bf16_kernel<<<(CC * CC + 255) / 256, 256, 0, stream>>>(Wo, wot, CC, CC);
    }

    // Stage 1: QKV projections
    {
        dim3 gq(MM / 128, CC / 64);    // (64, 9)
        dim3 gkv(MM / 128, KVC / 64);  // (64, 3)
        gemm_wmma_kernel<<<gq, 128, 0, stream>>>(xb, wqt, bq, qb, MM, CC,  CC, 0);
        gemm_wmma_kernel<<<gkv, 128, 0, stream>>>(xb, wkt, bk, kb, MM, KVC, CC, 0);
        gemm_wmma_kernel<<<gkv, 128, 0, stream>>>(xb, wvt, bv, vt, MM, KVC, CC, 1);
    }

    // Stage 2: causal GQA flash attention
    {
        dim3 ga(BB * NH, TT / 64);    // (72, 16)
        attn_wmma_kernel<<<ga, 128, 0, stream>>>(qb, kb, vt, yb);
    }

    // Stage 3: output projection (fp32 out + bias)
    {
        dim3 go(MM / 128, CC / 64);
        gemm_wmma_kernel<<<go, 128, 0, stream>>>(yb, wot, bo, (void*)out, MM, CC, CC, 2);
    }
}
